// Decay_29094108463320
// MI455X (gfx1250) — matrix operand order repaired and checked by reading, not compiled, not run on hardware
//
#include <hip/hip_runtime.h>

// ---------------------------------------------------------------------------
// MI455X / gfx1250 implementation of the gated decay-memory layer.
// Matrix path: bf16 V_WMMA_F32_16X16X32_BF16, f32 accumulation.
// __launch_bounds__(256) => 8 wave32/block, full VGPR budget, no spills.
// ---------------------------------------------------------------------------

typedef __attribute__((ext_vector_type(16))) __bf16 v16bf;
typedef __attribute__((ext_vector_type(8)))  __bf16 v8bf;
typedef __attribute__((ext_vector_type(8)))  float  v8f;

static constexpr int Bn = 4;
static constexpr int Sn = 4096;
static constexpr int En = 2048;
static constexpr int Mn = 2048;
static constexpr int BSn = Bn * Sn;          // 16384 rows
static constexpr int NCH = 8;                // scan chunks per sequence
static constexpr int CH  = Sn / NCH;         // 512
static constexpr int LDK = 40;               // padded LDS K-stride (80B rows)

__device__ __forceinline__ float sigmoidf_(float x) {
  return 1.0f / (1.0f + __expf(-x));
}

__device__ __forceinline__ v8f wmma_bf16(v16bf a, v16bf b, v8f c) {
  return __builtin_amdgcn_wmma_f32_16x16x32_bf16(
      /*neg_a=*/false, a, /*neg_b=*/false, b,
      /*c_mod=*/(short)0, c, /*reuse_a=*/false, /*reuse_b=*/false);
}

// A fragment (16x32 bf16): lane l<16 holds row l, K {0..7,16..23};
// lane l>=16 holds row l-16, K {8..15,24..31}.  Two 16B LDS loads.
__device__ __forceinline__ v16bf frag_a(const __bf16* t, int lane) {
  const __bf16* p = t + (size_t)(lane & 15) * LDK + ((lane >> 4) << 3);
  union { v16bf v; v8bf h[2]; } u;
  u.h[0] = *(const v8bf*)(p);
  u.h[1] = *(const v8bf*)(p + 16);
  return u.v;
}

// B fragment (32x16 bf16, LDS tile stored N-major [n][k]): lane l<16 holds
// col l, K {0..7,16..23}; lane l>=16 holds col l-16, K {8..15,24..31}.  Two 16B LDS loads.
__device__ __forceinline__ v16bf frag_b(const __bf16* t, int lane) {
  const __bf16* p = t + (size_t)(lane & 15) * LDK + ((lane >> 4) << 3);
  union { v16bf v; v8bf h[2]; } u;
  u.h[0] = *(const v8bf*)(p);
  u.h[1] = *(const v8bf*)(p + 16);
  return u.v;
}

// --------------------------- conversion kernels ----------------------------

__global__ void __launch_bounds__(256)
cvt_f32_to_bf16(const float* __restrict__ in, __bf16* __restrict__ out, int n) {
  int i = (blockIdx.x * blockDim.x + threadIdx.x) * 4;
  if (i + 3 < n) {
    float4 f = *(const float4*)(in + i);
    out[i + 0] = (__bf16)f.x;
    out[i + 1] = (__bf16)f.y;
    out[i + 2] = (__bf16)f.z;
    out[i + 3] = (__bf16)f.w;
  }
}

// in: R x C f32 (row-major) -> out: C x R bf16 (row-major), i.e. out[c][r]=in[r][c]
__global__ void __launch_bounds__(256)
cvt_transpose_bf16(const float* __restrict__ in, __bf16* __restrict__ out,
                   int R, int C) {
  __shared__ float tile[32][33];
  int c0 = blockIdx.x * 32, r0 = blockIdx.y * 32;
  int tx = threadIdx.x & 31, ty = threadIdx.x >> 5;   // 256 threads: ty 0..7
#pragma unroll
  for (int i = ty; i < 32; i += 8)
    tile[i][tx] = in[(size_t)(r0 + i) * C + (c0 + tx)];
  __syncthreads();
#pragma unroll
  for (int i = ty; i < 32; i += 8)
    out[(size_t)(c0 + i) * R + (r0 + tx)] = (__bf16)tile[tx][i];
}

// ----------------------- fused 3-way projection GEMM -----------------------
// Computes, for a 128x64 output tile:
//   g = sigmoid(x@Wg + bg); v = x@Wv; q = sigmoid(x@Wq + bq)
//   store = v*g*scale  -> f32 ws ;  que = q -> f32 ws
__global__ void __launch_bounds__(256)
proj_gemm(const __bf16* __restrict__ xbf,
          const __bf16* __restrict__ WgT,
          const __bf16* __restrict__ WvT,
          const __bf16* __restrict__ WqT,
          const float* __restrict__ bg,
          const float* __restrict__ bq,
          float* __restrict__ storef,
          float* __restrict__ quef) {
  __shared__ __bf16 sA [128 * LDK];
  __shared__ __bf16 sBg[ 64 * LDK];
  __shared__ __bf16 sBv[ 64 * LDK];
  __shared__ __bf16 sBq[ 64 * LDK];

  const int row0 = blockIdx.y * 128;
  const int col0 = blockIdx.x * 64;
  const int tid  = threadIdx.x;
  const int lane = tid & 31;
  const int wid  = tid >> 5;
  const int wr   = (wid & 3) * 32;   // wave row offset in tile
  const int wc   = (wid >> 2) * 32;  // wave col offset in tile

  const v8f zf = {0, 0, 0, 0, 0, 0, 0, 0};
  v8f ag[2][2], av[2][2], aq[2][2];
#pragma unroll
  for (int i = 0; i < 2; ++i)
#pragma unroll
    for (int j = 0; j < 2; ++j) { ag[i][j] = zf; av[i][j] = zf; aq[i][j] = zf; }

  const int arow = tid >> 1, ahalf = (tid & 1) * 16;   // A: 2x uint4 / thread
  const int bn = tid >> 2, bq4 = (tid & 3) * 8;        // B: 1x uint4 / thread

  for (int k0 = 0; k0 < En; k0 += 32) {
    __syncthreads();
    {
      const uint4* ga = (const uint4*)(xbf + (size_t)(row0 + arow) * En + k0 + ahalf);
      uint4 a0 = ga[0], a1 = ga[1];
      *(uint4*)&sA[arow * LDK + ahalf]     = a0;
      *(uint4*)&sA[arow * LDK + ahalf + 8] = a1;
      size_t boff = (size_t)(col0 + bn) * En + k0 + bq4;
      *(uint4*)&sBg[bn * LDK + bq4] = *(const uint4*)(WgT + boff);
      *(uint4*)&sBv[bn * LDK + bq4] = *(const uint4*)(WvT + boff);
      *(uint4*)&sBq[bn * LDK + bq4] = *(const uint4*)(WqT + boff);
    }
    __syncthreads();

    v16bf a0 = frag_a(&sA[(wr + 0)  * LDK], lane);
    v16bf a1 = frag_a(&sA[(wr + 16) * LDK], lane);

    v16bf b0 = frag_b(&sBg[(wc + 0)  * LDK], lane);
    v16bf b1 = frag_b(&sBg[(wc + 16) * LDK], lane);
    ag[0][0] = wmma_bf16(a0, b0, ag[0][0]);
    ag[0][1] = wmma_bf16(a0, b1, ag[0][1]);
    ag[1][0] = wmma_bf16(a1, b0, ag[1][0]);
    ag[1][1] = wmma_bf16(a1, b1, ag[1][1]);

    b0 = frag_b(&sBv[(wc + 0)  * LDK], lane);
    b1 = frag_b(&sBv[(wc + 16) * LDK], lane);
    av[0][0] = wmma_bf16(a0, b0, av[0][0]);
    av[0][1] = wmma_bf16(a0, b1, av[0][1]);
    av[1][0] = wmma_bf16(a1, b0, av[1][0]);
    av[1][1] = wmma_bf16(a1, b1, av[1][1]);

    b0 = frag_b(&sBq[(wc + 0)  * LDK], lane);
    b1 = frag_b(&sBq[(wc + 16) * LDK], lane);
    aq[0][0] = wmma_bf16(a0, b0, aq[0][0]);
    aq[0][1] = wmma_bf16(a0, b1, aq[0][1]);
    aq[1][0] = wmma_bf16(a1, b0, aq[1][0]);
    aq[1][1] = wmma_bf16(a1, b1, aq[1][1]);
  }

  const float SCALE = 0.22360679775f;  // sqrt(1 - 0.95)
#pragma unroll
  for (int i = 0; i < 2; ++i) {
#pragma unroll
    for (int j = 0; j < 2; ++j) {
      int n     = col0 + wc + j * 16 + (lane & 15);
      int rbase = row0 + wr + i * 16 + ((lane >> 4) << 3);
      float bgv = bg[n], bqv = bq[n];
#pragma unroll
      for (int e = 0; e < 8; ++e) {
        size_t off = (size_t)(rbase + e) * Mn + n;
        float g = sigmoidf_(ag[i][j][e] + bgv);
        storef[off] = av[i][j][e] * g * SCALE;
        quef[off]   = sigmoidf_(aq[i][j][e] + bqv);
      }
    }
  }
}

// ----------------------------- decay scan ----------------------------------
// Pass 1: chunk-local inclusive scan in place; emit per-chunk carries.
__global__ void __launch_bounds__(256)
scan_pass1(float* __restrict__ storef, float* __restrict__ carry) {
  const float DECAY = 0.95f;
  int t = blockIdx.x * blockDim.x + threadIdx.x;    // B*NCH*M threads
  int m = t % Mn;
  int rest = t / Mn;
  int c = rest % NCH;
  int b = rest / NCH;
  size_t base = ((size_t)b * Sn + (size_t)c * CH) * Mn + m;
  float mem = 0.0f;
  for (int s = 0; s < CH; ++s) {
    size_t idx = base + (size_t)s * Mn;
    mem = storef[idx] + DECAY * mem;
    storef[idx] = mem;
  }
  carry[((size_t)(b * NCH + c)) * Mn + m] = mem;
}

// Pass 2: fold decay^CH-weighted carry from previous chunks, apply que gate,
// emit bf16 `load` for the final GEMM.
__global__ void __launch_bounds__(256)
scan_pass2(const float* __restrict__ storef,
           const float* __restrict__ quef,
           const float* __restrict__ carry,
           __bf16* __restrict__ loadbf) {
  const float DECAY = 0.95f;
  const float SCALE = 0.22360679775f;
  int t = blockIdx.x * blockDim.x + threadIdx.x;
  int m = t % Mn;
  int rest = t / Mn;
  int c = rest % NCH;
  int b = rest / NCH;
  size_t base = ((size_t)b * Sn + (size_t)c * CH) * Mn + m;

  float Dch = __powf(DECAY, (float)CH);             // ~4e-12
  float cin = 0.0f;
  for (int j = 0; j < c; ++j)
    cin = carry[((size_t)(b * NCH + j)) * Mn + m] + Dch * cin;

  float p = DECAY;
  for (int s = 0; s < CH; ++s) {
    size_t idx = base + (size_t)s * Mn;
    float mem = storef[idx] + p * cin;
    p *= DECAY;
    loadbf[idx] = (__bf16)(mem * quef[idx] * SCALE);
  }
}

// ----------------------------- output GEMM ---------------------------------
// out(BS x E) = load(BS x M, bf16) @ WoT(E x M, bf16)^T  (WoT stored [e][m])
__global__ void __launch_bounds__(256)
out_gemm(const __bf16* __restrict__ loadbf,
         const __bf16* __restrict__ WoT,
         float* __restrict__ out) {
  __shared__ __bf16 sA[128 * LDK];
  __shared__ __bf16 sB[ 64 * LDK];

  const int row0 = blockIdx.y * 128;
  const int col0 = blockIdx.x * 64;
  const int tid  = threadIdx.x;
  const int lane = tid & 31;
  const int wid  = tid >> 5;
  const int wr   = (wid & 3) * 32;
  const int wc   = (wid >> 2) * 32;

  const v8f zf = {0, 0, 0, 0, 0, 0, 0, 0};
  v8f acc[2][2];
#pragma unroll
  for (int i = 0; i < 2; ++i)
#pragma unroll
    for (int j = 0; j < 2; ++j) acc[i][j] = zf;

  const int arow = tid >> 1, ahalf = (tid & 1) * 16;
  const int bn = tid >> 2, bq4 = (tid & 3) * 8;

  for (int k0 = 0; k0 < Mn; k0 += 32) {
    __syncthreads();
    {
      const uint4* ga = (const uint4*)(loadbf + (size_t)(row0 + arow) * Mn + k0 + ahalf);
      uint4 a0 = ga[0], a1 = ga[1];
      *(uint4*)&sA[arow * LDK + ahalf]     = a0;
      *(uint4*)&sA[arow * LDK + ahalf + 8] = a1;
      *(uint4*)&sB[bn * LDK + bq4] =
          *(const uint4*)(WoT + (size_t)(col0 + bn) * Mn + k0 + bq4);
    }
    __syncthreads();

    v16bf a0 = frag_a(&sA[(wr + 0)  * LDK], lane);
    v16bf a1 = frag_a(&sA[(wr + 16) * LDK], lane);
    v16bf b0 = frag_b(&sB[(wc + 0)  * LDK], lane);
    v16bf b1 = frag_b(&sB[(wc + 16) * LDK], lane);
    acc[0][0] = wmma_bf16(a0, b0, acc[0][0]);
    acc[0][1] = wmma_bf16(a0, b1, acc[0][1]);
    acc[1][0] = wmma_bf16(a1, b0, acc[1][0]);
    acc[1][1] = wmma_bf16(a1, b1, acc[1][1]);
  }

#pragma unroll
  for (int i = 0; i < 2; ++i) {
#pragma unroll
    for (int j = 0; j < 2; ++j) {
      int n     = col0 + wc + j * 16 + (lane & 15);
      int rbase = row0 + wr + i * 16 + ((lane >> 4) << 3);
#pragma unroll
      for (int e = 0; e < 8; ++e)
        out[(size_t)(rbase + e) * En + n] = acc[i][j][e];
    }
  }
}

// ------------------------------- launcher ----------------------------------

extern "C" void kernel_launch(void* const* d_in, const int* in_sizes, int n_in,
                              void* d_out, int out_size, void* d_ws, size_t ws_size,
                              hipStream_t stream) {
  (void)in_sizes; (void)n_in; (void)out_size; (void)ws_size;
  const float* x  = (const float*)d_in[0];
  const float* Wv = (const float*)d_in[1];
  const float* Wg = (const float*)d_in[2];
  const float* bg = (const float*)d_in[3];
  const float* Wq = (const float*)d_in[4];
  const float* bq = (const float*)d_in[5];
  const float* Wo = (const float*)d_in[6];
  float* out = (float*)d_out;

  // Workspace layout (all offsets 256B aligned)
  char* w = (char*)d_ws;
  __bf16* xbf = (__bf16*)w;              w += (size_t)BSn * En * 2;   // 64 MiB
  __bf16* WgT = (__bf16*)w;              w += (size_t)En * Mn * 2;    //  8 MiB
  __bf16* WvT = (__bf16*)w;              w += (size_t)En * Mn * 2;
  __bf16* WqT = (__bf16*)w;              w += (size_t)En * Mn * 2;
  __bf16* WoT = (__bf16*)w;              w += (size_t)Mn * En * 2;
  float*  storef = (float*)w;            w += (size_t)BSn * Mn * 4;   // 128 MiB
  float*  quef   = (float*)w;            w += (size_t)BSn * Mn * 4;   // 128 MiB
  __bf16* loadbf = (__bf16*)w;           w += (size_t)BSn * Mn * 2;   // 64 MiB
  float*  carry  = (float*)w;            w += (size_t)Bn * NCH * Mn * 4;

  // 1) x -> bf16
  {
    int n = BSn * En;
    cvt_f32_to_bf16<<<(n / 4 + 255) / 256, 256, 0, stream>>>(x, xbf, n);
  }
  // 2) weights -> bf16, K-contiguous transposed layouts
  cvt_transpose_bf16<<<dim3(Mn / 32, En / 32), 256, 0, stream>>>(Wg, WgT, En, Mn);
  cvt_transpose_bf16<<<dim3(Mn / 32, En / 32), 256, 0, stream>>>(Wv, WvT, En, Mn);
  cvt_transpose_bf16<<<dim3(Mn / 32, En / 32), 256, 0, stream>>>(Wq, WqT, En, Mn);
  cvt_transpose_bf16<<<dim3(En / 32, Mn / 32), 256, 0, stream>>>(Wo, WoT, Mn, En);

  // 3) fused projections (g, v, q) -> store, que
  proj_gemm<<<dim3(Mn / 64, BSn / 128), 256, 0, stream>>>(
      xbf, WgT, WvT, WqT, bg, bq, storef, quef);

  // 4) decay scan (chunked two-pass) -> load (bf16)
  {
    int nt = Bn * NCH * Mn;  // 65536
    scan_pass1<<<nt / 256, 256, 0, stream>>>(storef, carry);
    scan_pass2<<<nt / 256, 256, 0, stream>>>(storef, quef, carry, loadbf);
  }

  // 5) output GEMM -> d_out
  out_gemm<<<dim3(En / 64, BSn / 128), 256, 0, stream>>>(loadbf, WoT, out);
}
